// Head_1082331758773
// MI455X (gfx1250) — compile-verified, matrix operand order checked by reading, not run on hardware
//
#include <hip/hip_runtime.h>
#include <hip/hip_bf16.h>

typedef __attribute__((ext_vector_type(16))) __bf16       v16bf;
typedef __attribute__((ext_vector_type(8)))  float        v8f;
typedef __attribute__((ext_vector_type(4)))  unsigned int u32x4;

#define T_SEQ  4096
#define C_EMB  1024
#define D_HEAD 128
#define BATCH  4
#define SCALE  0.03125f   // C ** -0.5 = 1/sqrt(1024)

union Frag {
    v16bf  v;
    __bf16 h[16];
    u32x4  q[2];
};

__device__ inline __bf16 f2bf(float f) {
    unsigned u = __builtin_bit_cast(unsigned, f);
    unsigned r = (u + 0x7FFFu + ((u >> 16) & 1u)) >> 16;
    unsigned short s = (unsigned short)r;
    return __builtin_bit_cast(__bf16, s);
}

// Load a 16-bit A/B WMMA fragment: lo 8 bf16 from p0, hi 8 bf16 from p1.
__device__ inline v16bf ld_frag(const __bf16* p0, const __bf16* p1) {
    Frag f;
    f.q[0] = *(const u32x4*)p0;
    f.q[1] = *(const u32x4*)p1;
    return f.v;
}

// Async copy 16 bytes global -> LDS (ASYNCcnt-tracked).
__device__ inline void async_cp16(unsigned lds_addr, const void* gaddr) {
    asm volatile("global_load_async_to_lds_b128 %0, %1, off"
                 :: "v"(lds_addr), "v"(gaddr) : "memory");
}

// ---------------- elementwise f32 -> bf16 ----------------
__global__ void cvt_bf16_kernel(const float* __restrict__ in, __bf16* __restrict__ out, int n) {
    int i = blockIdx.x * blockDim.x + threadIdx.x;
    int stride = gridDim.x * blockDim.x;
    for (; i < n; i += stride) out[i] = f2bf(in[i]);
}

// ---------------- W [1024,128] f32 -> Wt [128,1024] bf16 ----------------
__global__ void transpose_w_kernel(const float* __restrict__ w, __bf16* __restrict__ wt) {
    int idx = blockIdx.x * blockDim.x + threadIdx.x;   // 131072 total
    int k = idx >> 7;
    int nn = idx & 127;
    wt[(size_t)nn * C_EMB + k] = f2bf(w[idx]);
}

// ---------------- projection GEMM: out = xbf @ Wt^T ----------------
// One wave per 16x16 output tile.  M = B*T = 16384, N = 128, K = 1024.
// vt_mode=1 writes V transposed as Vt[b][d][t].
__global__ void __launch_bounds__(128) proj_kernel(const __bf16* __restrict__ xbf,
                                                   const __bf16* __restrict__ wt,
                                                   __bf16* __restrict__ out, int vt_mode) {
    int wave = blockIdx.x * (blockDim.x >> 5) + (threadIdx.x >> 5);
    int lane = threadIdx.x & 31;
    int mt = wave >> 3;       // 0..1023  (16-row tile of M)
    int nt = wave & 7;        // 0..7     (16-col tile of N)
    int n  = lane & 15;
    int kb = (lane >> 4) * 8;

    const __bf16* arow = xbf + (size_t)(mt * 16 + n) * C_EMB;  // A row m = lane&15
    const __bf16* brow = wt  + (size_t)(nt * 16 + n) * C_EMB;  // B col n = lane&15

    v8f acc = {};
    for (int c = 0; c < 32; ++c) {
        const __bf16* pa = arow + 32 * c + kb;
        const __bf16* pb = brow + 32 * c + kb;
        v16bf af = ld_frag(pa, pa + 16);
        v16bf bf = ld_frag(pb, pb + 16);
        acc = __builtin_amdgcn_wmma_f32_16x16x32_bf16(false, af, false, bf,
                                                      (short)0, acc, false, false);
    }

    int mbase = (lane >> 4) * 8;
#pragma unroll
    for (int r = 0; r < 8; ++r) {
        int row = mt * 16 + mbase + r;
        int col = nt * 16 + n;
        __bf16 val = f2bf(acc[r]);
        if (!vt_mode) {
            out[(size_t)row * D_HEAD + col] = val;
        } else {
            int bb = row >> 12;              // batch
            int t  = row & (T_SEQ - 1);      // time
            out[((size_t)bb * D_HEAD + col) * T_SEQ + t] = val;
        }
    }
}

// ---------------- flash attention ----------------
// 4 waves / 128 threads per workgroup; workgroup owns 64 consecutive queries,
// each wave a 16-query tile.  K/V 32-key tiles are staged into LDS once per
// workgroup via async global->LDS copies and shared by all 4 waves.
__global__ void __launch_bounds__(128) attn_kernel(const __bf16* __restrict__ Q,
                                                   const __bf16* __restrict__ K,
                                                   const __bf16* __restrict__ Vt,
                                                   float* __restrict__ out) {
    __shared__ __bf16 kTile[32 * D_HEAD];     // [key 0..31][d 0..127]   8 KB
    __shared__ __bf16 vTile[D_HEAD * 32];     // [d 0..127][key 0..31]   8 KB
    __shared__ float  pT[4][16][32];          // per-wave P restage      8 KB

    int tid  = threadIdx.x;
    int wid  = tid >> 5;
    int lane = tid & 31;

    int b  = blockIdx.x >> 6;                 // 64 blocks per batch
    int q0 = (blockIdx.x & 63) << 6;          // 64-query workgroup tile
    int qw0 = q0 + wid * 16;                  // this wave's 16-query tile

    const __bf16* Qb = Q  + (size_t)b * T_SEQ * D_HEAD;
    const __bf16* Kb = K  + (size_t)b * T_SEQ * D_HEAD;
    const __bf16* Vb = Vt + (size_t)b * D_HEAD * T_SEQ;

    int n     = lane & 15;
    int kb    = (lane >> 4) * 8;
    int mbase = (lane >> 4) * 8;

    unsigned kBase = (unsigned)(size_t)kTile;
    unsigned vBase = (unsigned)(size_t)vTile;

    // Q fragments: 16 x 128 bf16 = 4 A-fragments, resident in VGPRs.
    v16bf qf[4];
    const __bf16* qrow = Qb + (size_t)(qw0 + n) * D_HEAD;
#pragma unroll
    for (int c = 0; c < 4; ++c) {
        const __bf16* p0 = qrow + 32 * c + kb;
        qf[c] = ld_frag(p0, p0 + 16);
    }

    v8f o[8] = {};
    float mrow[8], lrow[8];
#pragma unroll
    for (int r = 0; r < 8; ++r) { mrow[r] = -1e30f; lrow[r] = 0.0f; }

    int s_end_w  = ((qw0 + 15) >> 5) << 5;    // this wave's causal frontier tile
    int s_end_wg = ((q0 + 63) >> 5) << 5;     // workgroup frontier (last wave)

    for (int s0 = 0; s0 <= s_end_wg; s0 += 32) {
        // ---- stage K tile (32x128 bf16) and V tile (128x32 bf16) into LDS ----
#pragma unroll
        for (int i = 0; i < 4; ++i) {
            int chunk = tid + 128 * i;        // 512 chunks of 16B
            int krow = chunk >> 4;            // key row 0..31
            int kc   = chunk & 15;            // 16B column chunk
            async_cp16(kBase + chunk * 16,
                       Kb + (size_t)(s0 + krow) * D_HEAD + kc * 8);
        }
#pragma unroll
        for (int i = 0; i < 4; ++i) {
            int chunk = tid + 128 * i;
            int drow = chunk >> 2;            // d row 0..127
            int dc   = chunk & 3;             // 16B column chunk
            async_cp16(vBase + chunk * 16,
                       Vb + (size_t)drow * T_SEQ + s0 + dc * 8);
        }
        asm volatile("s_wait_asynccnt 0x0" ::: "memory");
        __syncthreads();

        if (s0 <= s_end_w) {                  // wave-uniform causal skip
            // ---- S = Q K^T for 32 keys (two 16x16 tiles), K from LDS ----
            v8f sa = {}, sb = {};
#pragma unroll
            for (int c = 0; c < 4; ++c) {
                const __bf16* k0 = kTile + (size_t)n * D_HEAD + 32 * c + kb;
                const __bf16* k1 = kTile + (size_t)(16 + n) * D_HEAD + 32 * c + kb;
                v16bf kf0 = ld_frag(k0, k0 + 16);
                v16bf kf1 = ld_frag(k1, k1 + 16);
                sa = __builtin_amdgcn_wmma_f32_16x16x32_bf16(false, qf[c], false, kf0,
                                                             (short)0, sa, false, false);
                sb = __builtin_amdgcn_wmma_f32_16x16x32_bf16(false, qf[c], false, kf1,
                                                             (short)0, sb, false, false);
            }

            // ---- scale + causal mask + online softmax ----
#pragma unroll
            for (int r = 0; r < 8; ++r) {
                int qg = qw0 + mbase + r;
                float a  = (s0 + n      <= qg) ? sa[r] * SCALE : -1e30f;
                float c2 = (s0 + 16 + n <= qg) ? sb[r] * SCALE : -1e30f;

                float mx = fmaxf(a, c2);
#pragma unroll
                for (int off = 1; off < 16; off <<= 1)
                    mx = fmaxf(mx, __shfl_xor(mx, off, 32));
                float mnew = fmaxf(mrow[r], mx);

                float p0 = __expf(a  - mnew);
                float p1 = __expf(c2 - mnew);
                float rs = p0 + p1;
#pragma unroll
                for (int off = 1; off < 16; off <<= 1)
                    rs += __shfl_xor(rs, off, 32);

                float alpha = __expf(mrow[r] - mnew);
                lrow[r] = lrow[r] * alpha + rs;
                mrow[r] = mnew;
#pragma unroll
                for (int t2 = 0; t2 < 8; ++t2) o[t2][r] *= alpha;

                pT[wid][mbase + r][n]      = p0;
                pT[wid][mbase + r][16 + n] = p1;
            }
            // wave-local LDS round trip: LDS ops are in-order per wave,
            // just wait for the stores to land (no workgroup barrier).
            asm volatile("s_wait_dscnt 0x0" ::: "memory");

            Frag pf;
#pragma unroll
            for (int j = 0; j < 8; ++j) {
                pf.h[j]     = f2bf(pT[wid][n][kb + j]);
                pf.h[8 + j] = f2bf(pT[wid][n][16 + kb + j]);
            }
            asm volatile("s_wait_dscnt 0x0" ::: "memory");

            // ---- O += P @ V (8 d-chunks of 16), V from LDS ----
#pragma unroll
            for (int t2 = 0; t2 < 8; ++t2) {
                const __bf16* v0 = vTile + (size_t)(t2 * 16 + n) * 32 + kb;
                v16bf vf = ld_frag(v0, v0 + 16);
                o[t2] = __builtin_amdgcn_wmma_f32_16x16x32_bf16(false, pf.v, false, vf,
                                                                (short)0, o[t2], false, false);
            }
        }
        __syncthreads();                      // protect K/V LDS before next stage
    }

    // ---- normalize and write ----
#pragma unroll
    for (int t2 = 0; t2 < 8; ++t2) {
#pragma unroll
        for (int r = 0; r < 8; ++r) {
            size_t row = (size_t)(b * T_SEQ + qw0 + mbase + r);
            out[row * D_HEAD + t2 * 16 + n] = o[t2][r] / lrow[r];
        }
    }
}

extern "C" void kernel_launch(void* const* d_in, const int* in_sizes, int n_in,
                              void* d_out, int out_size, void* d_ws, size_t ws_size,
                              hipStream_t stream) {
    const float* x  = (const float*)d_in[0];
    const float* Wk = (const float*)d_in[1];
    const float* Wq = (const float*)d_in[2];
    const float* Wv = (const float*)d_in[3];

    char* ws = (char*)d_ws;
    size_t off = 0;
    __bf16* xbf = (__bf16*)(ws + off); off += (size_t)BATCH * T_SEQ * C_EMB * 2;   // 32 MB
    __bf16* wkt = (__bf16*)(ws + off); off += (size_t)C_EMB * D_HEAD * 2;          // 256 KB
    __bf16* wqt = (__bf16*)(ws + off); off += (size_t)C_EMB * D_HEAD * 2;
    __bf16* wvt = (__bf16*)(ws + off); off += (size_t)C_EMB * D_HEAD * 2;
    __bf16* Qb  = (__bf16*)(ws + off); off += (size_t)BATCH * T_SEQ * D_HEAD * 2;  // 4 MB
    __bf16* Kb  = (__bf16*)(ws + off); off += (size_t)BATCH * T_SEQ * D_HEAD * 2;
    __bf16* Vt  = (__bf16*)(ws + off); off += (size_t)BATCH * T_SEQ * D_HEAD * 2;

    cvt_bf16_kernel<<<4096, 256, 0, stream>>>(x, xbf, BATCH * T_SEQ * C_EMB);
    transpose_w_kernel<<<512, 256, 0, stream>>>(Wq, wqt);
    transpose_w_kernel<<<512, 256, 0, stream>>>(Wk, wkt);
    transpose_w_kernel<<<512, 256, 0, stream>>>(Wv, wvt);

    // 1024 M-tiles * 8 N-tiles = 8192 waves, 4 waves/block
    proj_kernel<<<2048, 128, 0, stream>>>(xbf, wqt, Qb, 0);
    proj_kernel<<<2048, 128, 0, stream>>>(xbf, wkt, Kb, 0);
    proj_kernel<<<2048, 128, 0, stream>>>(xbf, wvt, Vt, 1);

    // 4 waves per block, 64 queries per block: B * T/64 = 256 blocks
    attn_kernel<<<BATCH * (T_SEQ / 64), 128, 0, stream>>>(Qb, Kb, Vt, (float*)d_out);
}
